// Network_15058155339963
// MI455X (gfx1250) — compile-verified, matrix operand order checked by reading, not run on hardware
//
// LiteFlowNet on MI455X (gfx1250): f16 WMMA implicit-GEMM conv engine + VALU glue.
// Precision: activations f16, WMMA f32 accumulate, flow fields f32.
// Conv tile: wave computes D(16 cout x 16 pixels) = A(weights, packed to the
// ISA 16-bit A-fragment layout) x B(im2col patch staged in LDS in fragment
// layout). Pixels ride the lane/N dimension -> coalesced NCHW stores.
// K-loop processes 64 K/iter (2 WMMAs) with double-buffered LDS staging;
// kernel is templated on (KH,KW) so all gather index math is div-free.
#include <hip/hip_runtime.h>
#include <hip/hip_bf16.h>
#include <math.h>

typedef _Float16 half_t;
typedef __attribute__((ext_vector_type(16))) _Float16 v16h;
typedef __attribute__((ext_vector_type(8)))  float    v8f;

#define BATCH 2
#define NTH 256

__device__ __forceinline__ float lrelu(float v){ return v >= 0.f ? v : 0.1f * v; }

// ---------------------------------------------------------------------------
// Weight packing: OIHW f32 -> per-wave A-fragment f16 layout
// [nTile][kTile][lane 0..31][half 0..15], lane: m=lane&15 (cout), g=lane>>4,
// half h<8 -> K=8g+h ; h>=8 -> K=16+8g+(h-8)   (ISA 16-bit A 16x32 layout)
// kTiles is rounded up to EVEN (K padded to multiple of 64 with zeros).
// ---------------------------------------------------------------------------
__global__ void k_pack_w(const float* __restrict__ w, half_t* __restrict__ pw,
                         int Cout, int Cin, int KH, int KW, int kTiles, long total){
  long p = (long)blockIdx.x * blockDim.x + threadIdx.x;
  if (p >= total) return;
  int h    = (int)(p & 15);
  int lane = (int)((p >> 4) & 31);
  long t   = p >> 9;
  int kt = (int)(t % kTiles);
  int nt = (int)(t / kTiles);
  int m = lane & 15, g = lane >> 4;
  int kk = (h < 8) ? (g * 8 + h) : (16 + g * 8 + (h - 8));
  int K  = kt * 32 + kk;
  int co = nt * 16 + m;
  int Kreal = Cin * KH * KW;
  half_t v = (half_t)0.f;
  if (K < Kreal && co < Cout){
    int cin = K / (KH * KW);
    int rem = K - cin * (KH * KW);
    int ky = rem / KW, kx = rem - ky * KW;
    v = (half_t)w[(((long)co * Cin + cin) * KH + ky) * KW + kx];
  }
  pw[p] = v;
}

// ---------------------------------------------------------------------------
// Implicit-GEMM conv + fused bias + LeakyReLU.  Block = 128 thr = 4 waves.
// grid.x = ceil(B*HO*WO/16) pixel tiles, grid.y = ceil(Cout/64) (4 waves split
// 64 couts, sharing the LDS-staged im2col patch). 64 K per iteration, double
// buffered: stage superchunk t+1 while the two WMMAs of superchunk t issue.
// ---------------------------------------------------------------------------
template<int KH, int KW>
__global__ void __launch_bounds__(128)
k_conv_wmma_t(const half_t* __restrict__ in, const half_t* __restrict__ pw,
              const float* __restrict__ bias, half_t* __restrict__ o16,
              float* __restrict__ o32, int relu,
              int Bn, int Cin, int H, int W, int Cout,
              int stride, int padh, int padw, int HO, int WO){
  constexpr int KHW = KH * KW;
  __shared__ __align__(16) half_t ldsB[2][2][32][16];
  __shared__ int  sIy[16], sIx[16], sRp[16], sB[16];
  __shared__ long sBase[16];
  const int tid = threadIdx.x;
  const int lane = tid & 31, wave = tid >> 5;
  const int pixBase = blockIdx.x * 16;
  const int coBase  = (blockIdx.y * 4 + wave) * 16;
  const int HWo = HO * WO;
  const int Mtot = Bn * HWo;
  const long HWi = (long)H * W;
  const int Kreal = Cin * KHW;
  const int kT2 = (Kreal + 63) >> 6;            // 64-K superchunks
  const bool active = (coBase < Cout);
  const half_t* pwT = pw + (long)(coBase >> 4) * (kT2 * 2) * 512;

  // one-time per-block pixel decomposition (keeps all div/mod out of K-loop)
  if (tid < 16){
    int mg = pixBase + tid;
    int b = 0, oy = 0, ox = 0, rp = 0;
    bool ok = (mg < Mtot);
    if (ok){ b = mg / HWo; rp = mg - b * HWo; oy = rp / WO; ox = rp - oy * WO; }
    sIy[tid] = ok ? (oy * stride - padh) : -(1 << 30);   // poisoned -> OOB
    sIx[tid] = ox * stride - padw;
    sRp[tid] = rp; sB[tid] = b;
    sBase[tid] = (long)b * Cin * HWi;
  }
  __syncthreads();

  // cooperative im2col gather of one 64xK x 16-pixel superchunk into
  // fragment layout (bijective 1024 slots; 8 elements per thread)
  auto stage = [&](int kt2, int buf){
    #pragma unroll
    for (int j = 0; j < 8; ++j){
      int idx = tid + 128 * j;
      int kk = idx >> 4, px = idx & 15;
      int K = kt2 * 64 + kk;
      half_t v = (half_t)0.f;
      if (K < Kreal){
        int cin = K / KHW;                 // constant divisor -> mul/shift
        int rem = K - cin * KHW;
        int ky = (KW == 1) ? rem : (rem / KW);
        int kx = (KW == 1) ? 0   : (rem - ky * KW);
        int iy = sIy[px] + ky;
        int ix = sIx[px] + kx;
        if ((unsigned)iy < (unsigned)H && (unsigned)ix < (unsigned)W)
          v = in[sBase[px] + (long)cin * HWi + (long)iy * W + ix];
      }
      int c2 = kk >> 5, kc = kk & 31;
      int ln = px + (((kc & 15) >= 8) ? 16 : 0);
      int hf = (kc & 7) + ((kc & 16) ? 8 : 0);
      ldsB[buf][c2][ln][hf] = v;
    }
  };

  stage(0, 0);
  __syncthreads();
  v8f acc = {};
  for (int kt2 = 0; kt2 < kT2; ++kt2){
    int cur = kt2 & 1;
    if (kt2 + 1 < kT2) stage(kt2 + 1, cur ^ 1);
    if (active){
      const half_t* pchunk = pwT + (long)kt2 * 1024;
      if (kt2 + 1 < kT2)
        __builtin_prefetch((const void*)(pchunk + 1024 + lane * 16), 0, 1);
      v16h b0 = *(const v16h*)(&ldsB[cur][0][lane][0]);
      v16h a0 = *(const v16h*)(pchunk + lane * 16);
      acc = __builtin_amdgcn_wmma_f32_16x16x32_f16(false, a0, false, b0,
                                                   (short)0, acc, false, false);
      v16h b1 = *(const v16h*)(&ldsB[cur][1][lane][0]);
      v16h a1 = *(const v16h*)(pchunk + 512 + lane * 16);
      acc = __builtin_amdgcn_wmma_f32_16x16x32_f16(false, a1, false, b1,
                                                   (short)0, acc, false, false);
    }
    __syncthreads();
  }

  if (!active) return;
  int px = lane & 15;
  int mg = pixBase + px;
  if (mg >= Mtot) return;
  int b = sB[px], rp = sRp[px];
  int coOff = (lane >> 4) * 8;
  #pragma unroll
  for (int r = 0; r < 8; ++r){
    int co = coBase + coOff + r;
    if (co < Cout){
      float v = acc[r] + bias[co];
      if (relu) v = lrelu(v);
      long o = ((long)b * Cout + co) * HWo + rp;
      if (o32) o32[o] = v; else o16[o] = (half_t)v;
    }
  }
}

// ----------------------------- glue kernels --------------------------------
__global__ void k_prep_image(const float* __restrict__ src, half_t* __restrict__ dst,
                             float m0, float m1, float m2, long chw, long hw, long n){
  long i = (long)blockIdx.x * blockDim.x + threadIdx.x; if (i >= n) return;
  long c = (i % chw) / hw;
  float m = (c == 0) ? m0 : ((c == 1) ? m1 : m2);
  dst[i] = (half_t)(src[i] - m);
}

__global__ void k_avgpool2(const half_t* __restrict__ in, half_t* __restrict__ out,
                           int BC, int H, int W, long n){
  long i = (long)blockIdx.x * blockDim.x + threadIdx.x; if (i >= n) return;
  int Ho = H >> 1, Wo = W >> 1;
  int x = (int)(i % Wo); long t = i / Wo;
  int y = (int)(t % Ho); long bc = t / Ho;
  const half_t* p = in + ((long)bc * H + 2 * y) * W + 2 * x;
  float v = ((float)p[0] + (float)p[1] + (float)p[W] + (float)p[W + 1]) * 0.25f;
  out[i] = (half_t)v;
}

__global__ void k_backwarp(const half_t* __restrict__ src, const float* __restrict__ flow,
                           float scale, half_t* __restrict__ dst,
                           int C, int H, int W, long n){
  long i = (long)blockIdx.x * blockDim.x + threadIdx.x; if (i >= n) return;
  long hw = (long)H * W;
  int x = (int)(i % W); long t = i / W;
  int y = (int)(t % H); long t2 = t / H;
  int c = (int)(t2 % C); int b = (int)(t2 / C);
  const float* fb = flow + (long)b * 2 * hw;
  long q = (long)y * W + x;
  float px = x + fb[q] * scale;
  float py = y + fb[hw + q] * scale;
  float x0f = floorf(px), y0f = floorf(py);
  int x0 = (int)x0f, y0 = (int)y0f;
  float wx = px - x0f, wy = py - y0f;
  const half_t* sp = src + ((long)b * C + c) * hw;
  float v = 0.f;
  #pragma unroll
  for (int dy = 0; dy < 2; ++dy)
    #pragma unroll
    for (int dx = 0; dx < 2; ++dx){
      int xi = x0 + dx, yi = y0 + dy;
      if ((unsigned)xi < (unsigned)W && (unsigned)yi < (unsigned)H){
        float wg = (dx ? wx : 1.f - wx) * (dy ? wy : 1.f - wy);
        v += wg * (float)sp[(long)yi * W + xi];
      }
    }
  dst[i] = (half_t)v;
}

// 49-way cost volume, mean over C, LeakyReLU fused, optional stride
__global__ void k_corr(const half_t* __restrict__ f1, const half_t* __restrict__ f2,
                       half_t* __restrict__ out, int C, int H, int W, int s, long n){
  long i = (long)blockIdx.x * blockDim.x + threadIdx.x; if (i >= n) return;
  int Hs = H / s, Ws = W / s;
  long hw = (long)H * W;
  int x = (int)(i % Ws); long t = i / Ws;
  int y = (int)(t % Hs); long t2 = t / Hs;
  int d = (int)(t2 % 49); int b = (int)(t2 / 49);
  int dy = d / 7 - 3, dx = d % 7 - 3;
  int y1 = y * s, x1 = x * s;
  int y2 = (y + dy) * s, x2 = (x + dx) * s;
  float v = 0.f;
  if ((unsigned)y2 < (unsigned)H && (unsigned)x2 < (unsigned)W){
    const half_t* p1 = f1 + (long)b * C * hw + (long)y1 * W + x1;
    const half_t* p2 = f2 + (long)b * C * hw + (long)y2 * W + x2;
    float a = 0.f;
    for (int c = 0; c < C; ++c) a += (float)p1[(long)c * hw] * (float)p2[(long)c * hw];
    v = a / (float)C;
  }
  out[i] = (half_t)lrelu(v);
}

// grouped 4x4 transposed conv, lhs_dilation=2, pad 2 (kernel pre-flip folded in)
__global__ void k_deconv_f32(const float* __restrict__ x, const float* __restrict__ w,
                             float* __restrict__ y, int C, int H, int W, long n){
  long i = (long)blockIdx.x * blockDim.x + threadIdx.x; if (i >= n) return;
  int Ho = 2 * H, Wo = 2 * W;
  int ox = (int)(i % Wo); long t = i / Wo;
  int oy = (int)(t % Ho); t /= Ho;
  int c = (int)(t % C); int b = (int)(t / C);
  const float* xb = x + ((long)b * C + c) * H * W;
  const float* wc = w + (long)c * 16;
  float acc = 0.f;
  for (int ky = 0; ky < 4; ++ky){
    if ((oy + ky) & 1) continue;
    int iy = (oy + ky - 2) >> 1; if ((unsigned)iy >= (unsigned)H) continue;
    for (int kx = 0; kx < 4; ++kx){
      if ((ox + kx) & 1) continue;
      int ix = (ox + kx - 2) >> 1; if ((unsigned)ix >= (unsigned)W) continue;
      acc += wc[(3 - ky) * 4 + (3 - kx)] * xb[(long)iy * W + ix];
    }
  }
  y[i] = acc;
}

__global__ void k_deconv_f16(const half_t* __restrict__ x, const float* __restrict__ w,
                             half_t* __restrict__ y, int C, int H, int W, long n){
  long i = (long)blockIdx.x * blockDim.x + threadIdx.x; if (i >= n) return;
  int Ho = 2 * H, Wo = 2 * W;
  int ox = (int)(i % Wo); long t = i / Wo;
  int oy = (int)(t % Ho); t /= Ho;
  int c = (int)(t % C); int b = (int)(t / C);
  const half_t* xb = x + ((long)b * C + c) * H * W;
  const float* wc = w + (long)c * 16;
  float acc = 0.f;
  for (int ky = 0; ky < 4; ++ky){
    if ((oy + ky) & 1) continue;
    int iy = (oy + ky - 2) >> 1; if ((unsigned)iy >= (unsigned)H) continue;
    for (int kx = 0; kx < 4; ++kx){
      if ((ox + kx) & 1) continue;
      int ix = (ox + kx - 2) >> 1; if ((unsigned)ix >= (unsigned)W) continue;
      acc += wc[(3 - ky) * 4 + (3 - kx)] * (float)xb[(long)iy * W + ix];
    }
  }
  y[i] = (half_t)acc;
}

__global__ void k_addf32(float* __restrict__ a, const float* __restrict__ b, long n){
  long i = (long)blockIdx.x * blockDim.x + threadIdx.x; if (i < n) a[i] += b[i];
}
__global__ void k_copyf32(float* __restrict__ a, const float* __restrict__ b, long n){
  long i = (long)blockIdx.x * blockDim.x + threadIdx.x; if (i < n) a[i] = b[i];
}
__global__ void k_scale_store(float* __restrict__ o, const float* __restrict__ f, long n){
  long i = (long)blockIdx.x * blockDim.x + threadIdx.x; if (i < n) o[i] = f[i] * 20.f;
}
__global__ void k_copy_ch(half_t* __restrict__ dst, const half_t* __restrict__ src,
                          int Cs, int Ct, int co, long hw, long n){
  long i = (long)blockIdx.x * blockDim.x + threadIdx.x; if (i >= n) return;
  long p = i % hw; long t = i / hw;
  int c = (int)(t % Cs); int b = (int)(t / Cs);
  dst[((long)b * Ct + co + c) * hw + p] = src[i];
}
__global__ void k_copy_ch_f32(half_t* __restrict__ dst, const float* __restrict__ src,
                              int Cs, int Ct, int co, long hw, long n){
  long i = (long)blockIdx.x * blockDim.x + threadIdx.x; if (i >= n) return;
  long p = i % hw; long t = i / hw;
  int c = (int)(t % Cs); int b = (int)(t / Cs);
  dst[((long)b * Ct + co + c) * hw + p] = (half_t)src[i];
}
__global__ void k_flow_mean(const float* __restrict__ flow, float* __restrict__ mean, long hw){
  __shared__ float red[256];
  int bc = blockIdx.x;
  const float* p = flow + (long)bc * hw;
  float a = 0.f;
  for (long i = threadIdx.x; i < hw; i += blockDim.x) a += p[i];
  red[threadIdx.x] = a; __syncthreads();
  for (int s = 128; s > 0; s >>= 1){
    if ((int)threadIdx.x < s) red[threadIdx.x] += red[threadIdx.x + s];
    __syncthreads();
  }
  if (threadIdx.x == 0) mean[bc] = red[0] / (float)hw;
}
// reg concat head: ch0 = ||im1 - warp(im2)||, ch1..2 = flow - mean(flow)
__global__ void k_reg_head(const half_t* __restrict__ im1, const half_t* __restrict__ im2w,
                           const float* __restrict__ flow, const float* __restrict__ mean,
                           half_t* __restrict__ dst, int Ct, int H, int W, long n){
  long i = (long)blockIdx.x * blockDim.x + threadIdx.x; if (i >= n) return;
  long hw = (long)H * W;
  long p = i % hw; int b = (int)(i / hw);
  float d = 0.f;
  for (int c = 0; c < 3; ++c){
    float t = (float)im1[((long)b * 3 + c) * hw + p] - (float)im2w[((long)b * 3 + c) * hw + p];
    d += t * t;
  }
  half_t* db = dst + (long)b * Ct * hw;
  db[p]          = (half_t)sqrtf(d);
  db[hw + p]     = (half_t)(flow[(long)b * 2 * hw + p]       - mean[b * 2 + 0]);
  db[2 * hw + p] = (half_t)(flow[((long)b * 2 + 1) * hw + p] - mean[b * 2 + 1]);
}
// dist = exp(-d^2 - max_ch(-d^2)),  div = 1/sum_ch(dist)
__global__ void k_dist(const float* __restrict__ d0, float* __restrict__ dist,
                       float* __restrict__ divb, int K2, long hw, long n){
  long i = (long)blockIdx.x * blockDim.x + threadIdx.x; if (i >= n) return;
  long p = i % hw; int b = (int)(i / hw);
  const float* s = d0 + (long)b * K2 * hw + p;
  float mx = -1e30f;
  for (int j = 0; j < K2; ++j){ float v = s[(long)j * hw]; v = -v * v; if (v > mx) mx = v; }
  float* dd = dist + (long)b * K2 * hw + p;
  float sum = 0.f;
  for (int j = 0; j < K2; ++j){
    float v = s[(long)j * hw];
    float e = __expf(-v * v - mx);
    dd[(long)j * hw] = e; sum += e;
  }
  divb[i] = 1.f / sum;
}
// sx = (sum_j sxw[j]*dist[j]*unfold(fx)[j] + sxb) * div  (same for y)
__global__ void k_reg_apply(const float* __restrict__ dist, const float* __restrict__ divb,
                            const float* __restrict__ flow,
                            const float* __restrict__ sxw, const float* __restrict__ sxb,
                            const float* __restrict__ syw, const float* __restrict__ syb,
                            float* __restrict__ out, int K2, int k, int H, int W, long n){
  long i = (long)blockIdx.x * blockDim.x + threadIdx.x; if (i >= n) return;
  long hw = (long)H * W;
  long p = i % hw; int b = (int)(i / hw);
  int y = (int)(p / W), x = (int)(p % W);
  int pd = (k - 1) / 2;
  float sx = sxb[0], sy = syb[0];
  const float* db = dist + (long)b * K2 * hw + p;
  const float* fx = flow + (long)b * 2 * hw;
  const float* fy = fx + hw;
  for (int j = 0; j < K2; ++j){
    int yy = y + j / k - pd, xx = x + j % k - pd;
    float ux = 0.f, uy = 0.f;
    if ((unsigned)yy < (unsigned)H && (unsigned)xx < (unsigned)W){
      long q = (long)yy * W + xx; ux = fx[q]; uy = fy[q];
    }
    float d = db[(long)j * hw];
    sx += sxw[j] * d * ux;
    sy += syw[j] * d * uy;
  }
  float dv = divb[i];
  out[(long)b * 2 * hw + p]       = sx * dv;
  out[((long)b * 2 + 1) * hw + p] = sy * dv;
}

// =========================== host orchestration ============================
struct Ctx { hipStream_t s; char* ws; size_t off; size_t cap; void* const* in; };
static inline void* balloc(Ctx& c, size_t bytes){
  void* p = c.ws + c.off;
  c.off = (c.off + bytes + 255) & ~((size_t)255);
  return p;
}
struct PW { const half_t* pw; const float* bias; int Cout, Cin, KH, KW; };

#define GRID1(n) dim3((unsigned)(((n) + NTH - 1) / NTH)), dim3(NTH), 0, ctx.s

static PW packW(Ctx& ctx, int wi, int bi, int Cout, int Cin, int KH, int KW){
  long kT = 2 * ((Cin * KH * KW + 63) / 64);     // pad K to multiple of 64
  long nT = (Cout + 15) / 16;
  long total = nT * kT * 512;
  half_t* pw = (half_t*)balloc(ctx, (size_t)total * sizeof(half_t));
  k_pack_w<<<GRID1(total)>>>((const float*)ctx.in[wi], pw, Cout, Cin, KH, KW, (int)kT, total);
  PW r; r.pw = pw; r.bias = (const float*)ctx.in[bi];
  r.Cout = Cout; r.Cin = Cin; r.KH = KH; r.KW = KW;
  return r;
}
static void runConv(Ctx& ctx, const PW& w, const half_t* in, void* out, int outF32,
                    int relu, int H, int W, int stride, int padh, int padw){
  int HO = (H + 2 * padh - w.KH) / stride + 1;
  int WO = (W + 2 * padw - w.KW) / stride + 1;
  long Mtot = (long)BATCH * HO * WO;
  dim3 g((unsigned)((Mtot + 15) / 16), (unsigned)((w.Cout + 63) / 64));
  half_t* p16 = outF32 ? nullptr : (half_t*)out;
  float*  p32 = outF32 ? (float*)out : nullptr;
#define CONV_CASE(kh, kw)                                                     \
  if (w.KH == kh && w.KW == kw){                                              \
    k_conv_wmma_t<kh, kw><<<g, dim3(128), 0, ctx.s>>>(in, w.pw, w.bias, p16,  \
        p32, relu, BATCH, w.Cin, H, W, w.Cout, stride, padh, padw, HO, WO);   \
    return; }
  CONV_CASE(1, 1) CONV_CASE(3, 3) CONV_CASE(5, 5) CONV_CASE(7, 7)
  CONV_CASE(5, 1) CONV_CASE(1, 5) CONV_CASE(7, 1) CONV_CASE(1, 7)
#undef CONV_CASE
}

// Feature-pyramid dims, level index 0..5 = (one..six)
static const int FH6[6] = {384, 192, 96, 48, 24, 12};
static const int FW6[6] = {768, 384, 192, 96, 48, 24};
static const int FC6[6] = {32, 32, 64, 96, 128, 192};

extern "C" void kernel_launch(void* const* d_in, const int* in_sizes, int n_in,
                              void* d_out, int out_size, void* d_ws, size_t ws_size,
                              hipStream_t stream){
  (void)in_sizes; (void)n_in;
  Ctx ctx; ctx.s = stream; ctx.ws = (char*)d_ws; ctx.off = 0; ctx.cap = ws_size; ctx.in = d_in;

  // Param indices assume JAX pytree flattening (sorted dict keys):
  // [0]=first [1]=second; feat 2..21; match 22..69; reg 70..171; sub 172..213.
  static const int feWi[10] = {5, 7, 9, 11, 13, 15, 17, 19, 21, 3};
  static const int feBi[10] = {4, 6, 8, 10, 12, 14, 16, 18, 20, 2};
  static const int feO[10]  = {32, 32, 32, 32, 64, 64, 96, 96, 128, 192};
  static const int feI[10]  = {3, 32, 32, 32, 32, 64, 64, 96, 96, 128};
  static const int feK[10]  = {7, 3, 3, 3, 3, 3, 3, 3, 3, 3};
  static const float LS5[5] = {10.f, 5.f, 2.5f, 1.25f, 0.625f};   // l=2..6
  static const int LK5[5]   = {7, 5, 5, 3, 3};
  static const int LK25[5]  = {49, 25, 25, 9, 9};
  static const int LSUB5[5] = {130, 130, 194, 258, 386};
  static const int LRM5[5]  = {131, 131, 131, 131, 195};
  static const int LRF5[3]  = {32, 64, 96};
  static const int Mm1b[5]  = {24, 34, 44, 53, 62};
  static const int Mup5[5]  = {32, 42, 52, 61, -1};
  static const int Mupc5[5] = {33, 43, -1, -1, -1};
  static const int Sm1b[5]  = {174, 182, 190, 198, 206};
  static const int Rb5[5]   = {70, 92, 114, 136, 154};

  // ---- persistent: packed feature weights + pyramids + flow/dist buffers
  PW few[10];
  for (int n = 0; n < 10; ++n)
    few[n] = packW(ctx, feWi[n], feBi[n], feO[n], feI[n], feK[n], feK[n]);

  half_t *img1[6], *img2[6], *fA[6], *fB[6];
  for (int i = 0; i < 6; ++i){
    size_t ihw = (size_t)FH6[i] * FW6[i];
    img1[i] = (half_t*)balloc(ctx, (size_t)BATCH * 3 * ihw * 2);
    img2[i] = (half_t*)balloc(ctx, (size_t)BATCH * 3 * ihw * 2);
    fA[i]   = (half_t*)balloc(ctx, (size_t)BATCH * FC6[i] * ihw * 2);
    fB[i]   = (half_t*)balloc(ctx, (size_t)BATCH * FC6[i] * ihw * 2);
  }
  long hwMax = (long)192 * 384;
  float* flowX = (float*)balloc(ctx, (size_t)BATCH * 2 * hwMax * 4);
  float* flowY = (float*)balloc(ctx, (size_t)BATCH * 2 * hwMax * 4);
  float* delta = (float*)balloc(ctx, (size_t)BATCH * 2 * hwMax * 4);
  float* meanB = (float*)balloc(ctx, 64 * sizeof(float));
  float* d0Buf = (float*)balloc(ctx, (size_t)BATCH * 49 * hwMax * 4);
  float* distB = (float*)balloc(ctx, (size_t)BATCH * 49 * hwMax * 4);
  float* divB  = (float*)balloc(ctx, (size_t)BATCH * hwMax * 4);
  size_t ck = ctx.off;   // scratch checkpoint: everything after is per-phase

  // ---- mean-subtract images (f16) and build 6-level pyramids
  {
    long hw0 = (long)FH6[0] * FW6[0];
    long chw = 3 * hw0;
    long n = (long)BATCH * chw;
    k_prep_image<<<GRID1(n)>>>((const float*)d_in[0], img1[0],
                               0.411618f, 0.434631f, 0.454253f, chw, hw0, n);
    k_prep_image<<<GRID1(n)>>>((const float*)d_in[1], img2[0],
                               0.410782f, 0.433645f, 0.452793f, chw, hw0, n);
    for (int i = 1; i < 6; ++i){
      long no = (long)BATCH * 3 * FH6[i] * FW6[i];
      k_avgpool2<<<GRID1(no)>>>(img1[i - 1], img1[i], BATCH * 3, FH6[i - 1], FW6[i - 1], no);
      k_avgpool2<<<GRID1(no)>>>(img2[i - 1], img2[i], BATCH * 3, FH6[i - 1], FW6[i - 1], no);
    }
  }

  // ---- feature extraction (all convs through the WMMA engine)
  for (int im = 0; im < 2; ++im){
    half_t** fe = im ? fB : fA;
    const half_t* ip = im ? img2[0] : img1[0];
    ctx.off = ck;
    half_t* tA = (half_t*)balloc(ctx, (size_t)BATCH * 64 * 192 * 384 * 2);
    half_t* tB = (half_t*)balloc(ctx, (size_t)BATCH * 64 * 192 * 384 * 2);
    runConv(ctx, few[0], ip,    fe[0], 0, 1, 384, 768, 1, 3, 3);
    runConv(ctx, few[1], fe[0], tA,    0, 1, 384, 768, 2, 1, 1);
    runConv(ctx, few[2], tA,    tB,    0, 1, 192, 384, 1, 1, 1);
    runConv(ctx, few[3], tB,    fe[1], 0, 1, 192, 384, 1, 1, 1);
    runConv(ctx, few[4], fe[1], tA,    0, 1, 192, 384, 2, 1, 1);
    runConv(ctx, few[5], tA,    fe[2], 0, 1, 96, 192, 1, 1, 1);
    runConv(ctx, few[6], fe[2], tA,    0, 1, 96, 192, 2, 1, 1);
    runConv(ctx, few[7], tA,    fe[3], 0, 1, 48, 96, 1, 1, 1);
    runConv(ctx, few[8], fe[3], fe[4], 0, 1, 48, 96, 2, 1, 1);
    runConv(ctx, few[9], fe[4], fe[5], 0, 1, 24, 48, 2, 1, 1);
  }

  // ---- coarse-to-fine flow pyramid
  float* flowCur = nullptr;
  for (int l = 6; l >= 2; --l){
    int li = l - 2, fi = l - 1;
    int H = FH6[fi], W = FW6[fi], C = FC6[fi];
    long hw = (long)H * W;
    float sc = LS5[li];
    int k = LK5[li], k2 = LK25[li], pd = (k - 1) / 2;
    ctx.off = ck;

    // per-level packed weights (recomputed each call: deterministic)
    int mb = Mm1b[li];
    PW m1 = packW(ctx, mb + 1, mb + 0, 128, 49, 3, 3);
    PW m2 = packW(ctx, mb + 3, mb + 2, 64, 128, 3, 3);
    PW m3 = packW(ctx, mb + 5, mb + 4, 32, 64, 3, 3);
    PW m4 = packW(ctx, mb + 7, mb + 6, 2, 32, k, k);
    int sb = Sm1b[li];
    PW s1 = packW(ctx, sb + 1, sb + 0, 128, LSUB5[li], 3, 3);
    PW s2 = packW(ctx, sb + 3, sb + 2, 64, 128, 3, 3);
    PW s3 = packW(ctx, sb + 5, sb + 4, 32, 64, 3, 3);
    PW s4 = packW(ctx, sb + 7, sb + 6, 2, 32, k, k);
    int rb = Rb5[li];
    int rm = rb + ((l < 5) ? 6 : 2);
    PW r1 = packW(ctx, rm + 1,  rm + 0,  128, LRM5[li], 3, 3);
    PW r2 = packW(ctx, rm + 3,  rm + 2,  128, 128, 3, 3);
    PW r3 = packW(ctx, rm + 5,  rm + 4,  64, 128, 3, 3);
    PW r4 = packW(ctx, rm + 7,  rm + 6,  64, 64, 3, 3);
    PW r5 = packW(ctx, rm + 9,  rm + 8,  32, 64, 3, 3);
    PW r6 = packW(ctx, rm + 11, rm + 10, 32, 32, 3, 3);
    PW rd1 = (l >= 5) ? packW(ctx, rb + 1, rb + 0, k2, 32, k, k)
                      : packW(ctx, rb + 1, rb + 0, k2, 32, k, 1);
    PW rd2 = rd1;
    if (l < 5) rd2 = packW(ctx, rb + 3, rb + 2, k2, k2, 1, k);
    const float* sxb = (const float*)d_in[rb + ((l < 5) ? 18 : 14)];
    const float* sxw = (const float*)d_in[rb + ((l < 5) ? 19 : 15)];
    const float* syb = (const float*)d_in[rb + ((l < 5) ? 20 : 16)];
    const float* syw = (const float*)d_in[rb + ((l < 5) ? 21 : 17)];

    float* work = (flowCur == flowX) ? flowY : flowX;
    float* fout = (work == flowX) ? flowY : flowX;

    half_t* tA  = (half_t*)balloc(ctx, (size_t)BATCH * 128 * hw * 2);
    half_t* tB2 = (half_t*)balloc(ctx, (size_t)BATCH * 128 * hw * 2);
    long nf = (long)BATCH * 2 * hw;

    // ===== matching =====
    const half_t *mf1 = fA[fi], *mf2 = fB[fi];
    int Cm = C;
    if (l == 2){
      PW mfw = packW(ctx, 23, 22, 64, 32, 1, 1);
      half_t* b1 = (half_t*)balloc(ctx, (size_t)BATCH * 64 * hw * 2);
      half_t* b2 = (half_t*)balloc(ctx, (size_t)BATCH * 64 * hw * 2);
      runConv(ctx, mfw, fA[fi], b1, 0, 1, H, W, 1, 0, 0);
      runConv(ctx, mfw, fB[fi], b2, 0, 1, H, W, 1, 0, 0);
      mf1 = b1; mf2 = b2; Cm = 64;
    }
    const half_t* f2m = mf2;
    float* flowU = nullptr;
    if (flowCur){
      flowU = work;
      k_deconv_f32<<<GRID1(nf)>>>(flowCur, (const float*)d_in[Mup5[li]],
                                  flowU, 2, H / 2, W / 2, nf);
      half_t* wb = (half_t*)balloc(ctx, (size_t)BATCH * Cm * hw * 2);
      long nw = (long)BATCH * Cm * hw;
      k_backwarp<<<GRID1(nw)>>>(mf2, flowU, sc, wb, Cm, H, W, nw);
      f2m = wb;
    }
    half_t* corr = (half_t*)balloc(ctx, (size_t)BATCH * 49 * hw * 2);
    if (l >= 4){
      long nc = (long)BATCH * 49 * hw;
      k_corr<<<GRID1(nc)>>>(mf1, f2m, corr, Cm, H, W, 1, nc);
    } else {
      half_t* corrS = (half_t*)balloc(ctx, (size_t)BATCH * 49 * (hw / 4) * 2);
      long ncs = (long)BATCH * 49 * (hw / 4);
      k_corr<<<GRID1(ncs)>>>(mf1, f2m, corrS, Cm, H, W, 2, ncs);
      long nc = (long)BATCH * 49 * hw;
      k_deconv_f16<<<GRID1(nc)>>>(corrS, (const float*)d_in[Mupc5[li]],
                                  corr, 49, H / 2, W / 2, nc);
    }
    runConv(ctx, m1, corr, tA, 0, 1, H, W, 1, 1, 1);
    runConv(ctx, m2, tA, tB2, 0, 1, H, W, 1, 1, 1);
    runConv(ctx, m3, tB2, tA, 0, 1, H, W, 1, 1, 1);
    runConv(ctx, m4, tA, delta, 1, 0, H, W, 1, pd, pd);
    if (flowU) k_addf32<<<GRID1(nf)>>>(flowU, delta, nf);
    else       k_copyf32<<<GRID1(nf)>>>(work, delta, nf);
    float* flowM = work;

    // ===== subpixel =====
    const half_t *sf1 = fA[fi], *sf2 = fB[fi];
    int Cs = C;
    if (l == 2){
      PW sfw = packW(ctx, 173, 172, 64, 32, 1, 1);
      half_t* b1 = (half_t*)balloc(ctx, (size_t)BATCH * 64 * hw * 2);
      half_t* b2 = (half_t*)balloc(ctx, (size_t)BATCH * 64 * hw * 2);
      runConv(ctx, sfw, fA[fi], b1, 0, 1, H, W, 1, 0, 0);
      runConv(ctx, sfw, fB[fi], b2, 0, 1, H, W, 1, 0, 0);
      sf1 = b1; sf2 = b2; Cs = 64;
    }
    half_t* w2 = (half_t*)balloc(ctx, (size_t)BATCH * Cs * hw * 2);
    long nw2 = (long)BATCH * Cs * hw;
    k_backwarp<<<GRID1(nw2)>>>(sf2, flowM, sc, w2, Cs, H, W, nw2);
    int Csub = LSUB5[li];
    half_t* cat = (half_t*)balloc(ctx, (size_t)BATCH * Csub * hw * 2);
    k_copy_ch<<<GRID1(nw2)>>>(cat, sf1, Cs, Csub, 0, hw, nw2);
    k_copy_ch<<<GRID1(nw2)>>>(cat, w2, Cs, Csub, Cs, hw, nw2);
    k_copy_ch_f32<<<GRID1(nf)>>>(cat, flowM, 2, Csub, 2 * Cs, hw, nf);
    runConv(ctx, s1, cat, tA, 0, 1, H, W, 1, 1, 1);
    runConv(ctx, s2, tA, tB2, 0, 1, H, W, 1, 1, 1);
    runConv(ctx, s3, tB2, tA, 0, 1, H, W, 1, 1, 1);
    runConv(ctx, s4, tA, delta, 1, 0, H, W, 1, pd, pd);
    k_addf32<<<GRID1(nf)>>>(flowM, delta, nf);
    float* flowS = flowM;

    // ===== regularization =====
    half_t* imw = (half_t*)balloc(ctx, (size_t)BATCH * 3 * hw * 2);
    long ni = (long)BATCH * 3 * hw;
    k_backwarp<<<GRID1(ni)>>>(img2[fi], flowS, sc, imw, 3, H, W, ni);
    k_flow_mean<<<dim3(BATCH * 2), dim3(256), 0, ctx.s>>>(flowS, meanB, hw);
    const half_t* rf = fA[fi];
    int Cr = C;
    if (l < 5){
      PW rfw = packW(ctx, rb + 5, rb + 4, 128, LRF5[li], 1, 1);
      half_t* rbuf = (half_t*)balloc(ctx, (size_t)BATCH * 128 * hw * 2);
      runConv(ctx, rfw, fA[fi], rbuf, 0, 1, H, W, 1, 0, 0);
      rf = rbuf; Cr = 128;
    }
    int Crm = LRM5[li];
    half_t* rcat = (half_t*)balloc(ctx, (size_t)BATCH * Crm * hw * 2);
    long nh = (long)BATCH * hw;
    k_reg_head<<<GRID1(nh)>>>(img1[fi], imw, flowS, meanB, rcat, Crm, H, W, nh);
    long nrc = (long)BATCH * Cr * hw;
    k_copy_ch<<<GRID1(nrc)>>>(rcat, rf, Cr, Crm, 3, hw, nrc);
    runConv(ctx, r1, rcat, tA, 0, 1, H, W, 1, 1, 1);
    runConv(ctx, r2, tA, tB2, 0, 1, H, W, 1, 1, 1);
    runConv(ctx, r3, tB2, tA, 0, 1, H, W, 1, 1, 1);
    runConv(ctx, r4, tA, tB2, 0, 1, H, W, 1, 1, 1);
    runConv(ctx, r5, tB2, tA, 0, 1, H, W, 1, 1, 1);
    runConv(ctx, r6, tA, tB2, 0, 1, H, W, 1, 1, 1);
    if (l >= 5){
      runConv(ctx, rd1, tB2, d0Buf, 1, 0, H, W, 1, pd, pd);
    } else {
      half_t* td = (half_t*)balloc(ctx, (size_t)BATCH * k2 * hw * 2);
      runConv(ctx, rd1, tB2, td, 0, 0, H, W, 1, pd, 0);
      runConv(ctx, rd2, td, d0Buf, 1, 0, H, W, 1, 0, pd);
    }
    k_dist<<<GRID1(nh)>>>(d0Buf, distB, divB, k2, hw, nh);
    k_reg_apply<<<GRID1(nh)>>>(distB, divB, flowS, sxw, sxb, syw, syb,
                               fout, k2, k, H, W, nh);
    flowCur = fout;
  }

  long no = (long)out_size;
  k_scale_store<<<GRID1(no)>>>((float*)d_out, flowCur, no);
}
